// ContinuousInvariantFeatureODE_30605936951524
// MI455X (gfx1250) — matrix operand order repaired and checked by reading, not compiled, not run on hardware
//
#include <hip/hip_runtime.h>
#include <math.h>

// ---------------------------------------------------------------------------
// CDNA5 (gfx1250) fused GNN message-passing layer.
// All GEMMs use v_wmma_f32_16x16x32_f16 (wave32 WMMA).
// Each wave owns TWO 16-row M-tiles so every B (weight) fragment fetched from
// L2 feeds two WMMAs -> halves vmem per matrix op.
// ---------------------------------------------------------------------------

typedef __attribute__((ext_vector_type(16))) _Float16 v16h;
typedef __attribute__((ext_vector_type(8)))  float    v8f;
typedef _Float16 half_t;

#define C_DIM   64
#define MC_DIM  128
#define KP1     224   // 3*C+4 = 196 padded to multiple of 32
#define UPD_KP  256   // C + MC + C = 256
#define TILE_M  64    // rows (edges / nodes) per workgroup
#define NTHREAD 64    // 2 waves, 32 rows each (2 M-tiles per wave)

__device__ __forceinline__ float gelu_f(float x) {
  return 0.5f * x * (1.0f + erff(x * 0.70710678118654752f));
}

__device__ __forceinline__ void atom_add_f32(float* p, float v) {
  __hip_atomic_fetch_add(p, v, __ATOMIC_RELAXED, __HIP_MEMORY_SCOPE_AGENT);
}

union frag_u { float4 f[2]; v16h h; };

// A fragment (16x32 f16): lanes 0-15 rows M, halves 0..7 = K kb..kb+7,
// halves 8..15 = K kb+16..kb+23, kb = k0 + 8*(lane>=16).
__device__ __forceinline__ v16h ld_fragA(const half_t* aW, int Sa, int k0, int lane) {
  const int r  = lane & 15;
  const int kb = k0 + ((lane >> 4) << 3);
  frag_u u;
  u.f[0] = *reinterpret_cast<const float4*>(aW + (size_t)r * Sa + kb);
  u.f[1] = *reinterpret_cast<const float4*>(aW + (size_t)r * Sa + kb + 16);
  return u.h;
}

// B fragment (32x16 f16): lane holds column n; halves 0..7 = K kb..kb+7,
// halves 8..15 = K kb+16..kb+23, kb = k0 + 8*(lane>=16) (from W^T [n][K]).
__device__ __forceinline__ v16h ld_fragB(const half_t* wt, int Kp, int n0, int k0, int lane) {
  const int nn = n0 + (lane & 15);
  const int kb = k0 + ((lane >> 4) << 3);
  const float4* p = reinterpret_cast<const float4*>(wt + (size_t)nn * Kp + kb);
  frag_u u;
  u.f[0] = p[0];
  u.f[1] = p[2];
  return u.h;
}

// Two 16-row M-tiles (32 rows) per wave: out = act(A * W^T + bias), fp16 out.
// One B fragment load feeds two WMMAs.
__device__ __forceinline__ void gemm_tile2(const half_t* aW, int Sa, int Kp,
                                           const half_t* wt, const float* bias,
                                           half_t* oW, int So, int Nout,
                                           bool act, int lane) {
  const int ks = Kp >> 5;
  const int r = lane & 15, hi = lane >> 4;
  for (int nt = 0; nt < (Nout >> 4); ++nt) {
    v8f acc0 = {0.f, 0.f, 0.f, 0.f, 0.f, 0.f, 0.f, 0.f};
    v8f acc1 = {0.f, 0.f, 0.f, 0.f, 0.f, 0.f, 0.f, 0.f};
    for (int k = 0; k < ks; ++k) {
      v16h b  = ld_fragB(wt, Kp, nt << 4, k << 5, lane);
      v16h a0 = ld_fragA(aW, Sa, k << 5, lane);
      v16h a1 = ld_fragA(aW + (size_t)16 * Sa, Sa, k << 5, lane);
      acc0 = __builtin_amdgcn_wmma_f32_16x16x32_f16(false, a0, false, b,
                                                    (short)0, acc0, false, false);
      acc1 = __builtin_amdgcn_wmma_f32_16x16x32_f16(false, a1, false, b,
                                                    (short)0, acc1, false, false);
    }
    const int n  = (nt << 4) + r;
    const float bv = bias[n];
#pragma unroll
    for (int v = 0; v < 8; ++v) {
      float x0 = acc0[v] + bv;
      float x1 = acc1[v] + bv;
      if (act) { x0 = gelu_f(x0); x1 = gelu_f(x1); }
      oW[(size_t)(v + (hi << 3)) * So + n]      = (half_t)x0;
      oW[(size_t)(16 + v + (hi << 3)) * So + n] = (half_t)x1;
    }
  }
}

// ---------------------------------------------------------------------------
// Weight prep: fp32 [K][N] -> fp16 transposed padded [N][Kpad]
// ---------------------------------------------------------------------------
__global__ void wprep_kernel(const float* __restrict__ w, half_t* __restrict__ wt,
                             int K, int Ncols, int Kpad) {
  int i = blockIdx.x * 256 + threadIdx.x;
  if (i >= Ncols * Kpad) return;
  int n = i / Kpad, k = i - n * Kpad;
  wt[(size_t)n * Kpad + k] = (k < K) ? (half_t)w[(size_t)k * Ncols + n] : (half_t)0.f;
}

// ---------------------------------------------------------------------------
// flow_dir normalization (3 elements)
// ---------------------------------------------------------------------------
__global__ void unorm_kernel(const float* __restrict__ fd, float* __restrict__ u) {
  float x = fd[0], y = fd[1], z = fd[2];
  float n = sqrtf(x * x + y * y + z * z) + 1e-8f;
  u[0] = x / n; u[1] = y / n; u[2] = z / n;
}

// ---------------------------------------------------------------------------
// LayerNorm -> fp16 feat, plus channel-sum partials for the global mean
// ---------------------------------------------------------------------------
__global__ __launch_bounds__(256) void ln_kernel(const float* __restrict__ h,
                                                 const float* __restrict__ g,
                                                 const float* __restrict__ b,
                                                 half_t* __restrict__ feat,
                                                 float* __restrict__ fsum, int N) {
  __shared__ float ps[C_DIM];
  int tid = threadIdx.x;
  if (tid < C_DIM) ps[tid] = 0.f;
  __syncthreads();
  int n = blockIdx.x * 256 + tid;
  if (n < N) {
    const float* x = h + (size_t)n * C_DIM;
    float mu = 0.f;
    for (int c = 0; c < C_DIM; ++c) mu += x[c];
    mu *= (1.0f / C_DIM);
    float var = 0.f;
    for (int c = 0; c < C_DIM; ++c) { float d = x[c] - mu; var += d * d; }
    var *= (1.0f / C_DIM);
    float inv = rsqrtf(var + 1e-5f);
    for (int c = 0; c < C_DIM; ++c) {
      float v = (x[c] - mu) * inv * g[c] + b[c];
      feat[(size_t)n * C_DIM + c] = (half_t)v;
      atomicAdd(&ps[c], v);            // LDS ds_add_f32
    }
  }
  __syncthreads();
  if (tid < C_DIM) atom_add_f32(&fsum[tid], ps[tid]);
}

// ---------------------------------------------------------------------------
// Tiny global-context MLP (64->64->64->64), one block of 64 threads
// ---------------------------------------------------------------------------
__global__ void global_mlp_kernel(const float* __restrict__ fsum, int N,
                                  const float* w1, const float* b1,
                                  const float* w2, const float* b2,
                                  const float* w3, const float* b3,
                                  half_t* __restrict__ gctx) {
  __shared__ float a[C_DIM], bb[C_DIM];
  int t = threadIdx.x;
  a[t] = fsum[t] * (1.0f / (float)N);
  __syncthreads();
  float s = 0.f;
  for (int k = 0; k < C_DIM; ++k) s += a[k] * w1[k * C_DIM + t];
  bb[t] = gelu_f(s + b1[t]);
  __syncthreads();
  s = 0.f;
  for (int k = 0; k < C_DIM; ++k) s += bb[k] * w2[k * C_DIM + t];
  a[t] = gelu_f(s + b2[t]);
  __syncthreads();
  s = 0.f;
  for (int k = 0; k < C_DIM; ++k) s += a[k] * w3[k * C_DIM + t];
  gctx[t] = (half_t)(s + b3[t]);
}

// ---------------------------------------------------------------------------
// Edge kernel: 64 edges per workgroup (2 waves x 32 rows);
// fused edge MLP + gate + atomic scatter.
// ---------------------------------------------------------------------------
__global__ __launch_bounds__(NTHREAD) void edge_kernel(
    const int* __restrict__ rows, const int* __restrict__ cols,
    const float* __restrict__ coords, const half_t* __restrict__ feat,
    const float* __restrict__ uvec,
    const half_t* __restrict__ ew1t, const float* __restrict__ eb1,
    const half_t* __restrict__ ew2t, const float* __restrict__ eb2,
    const half_t* __restrict__ ew3t, const float* __restrict__ eb3,
    const half_t* __restrict__ gw1t, const float* __restrict__ gb1,
    const float* __restrict__ gw2, const float* __restrict__ gb2,
    float* __restrict__ agg, float* __restrict__ deg, int E) {
  __shared__ __align__(16) half_t sA[TILE_M * KP1];      // edge_input, later h2
  __shared__ __align__(16) half_t sB[TILE_M * MC_DIM];   // h1, later h3
  __shared__ float sGate[TILE_M];
  __shared__ int sRow[TILE_M];
  __shared__ int sCol[TILE_M];

  const int tid = threadIdx.x;
  const int e0 = blockIdx.x * TILE_M;

  {
    int e = e0 + tid;
    sRow[tid] = (e < E) ? rows[e] : -1;
    sCol[tid] = (e < E) ? cols[e] : -1;
  }
  __syncthreads();

  // Build edge_input tile [64 x 224] fp16: [fi | fj | fi-fj | 4 scalars | pad]
  for (int i = tid; i < TILE_M * C_DIM; i += NTHREAD) {
    int e = i >> 6, c = i & 63;
    int r = sRow[e], q = sCol[e];
    float fi = 0.f, fj = 0.f;
    if (r >= 0) {
      fi = (float)feat[(size_t)r * C_DIM + c];
      fj = (float)feat[(size_t)q * C_DIM + c];
    }
    half_t* rp = sA + (size_t)e * KP1;
    rp[c] = (half_t)fi;
    rp[C_DIM + c] = (half_t)fj;
    rp[2 * C_DIM + c] = (half_t)(fi - fj);
  }
  {
    int e = tid, r = sRow[e], q = sCol[e];
    float s0 = 0.f, s1 = 0.f, s2 = 0.f, s3 = 0.f;
    if (r >= 0) {
      float xi0 = coords[r * 3], xi1 = coords[r * 3 + 1], xi2 = coords[r * 3 + 2];
      float xj0 = coords[q * 3], xj1 = coords[q * 3 + 1], xj2 = coords[q * 3 + 2];
      float r0 = xi0 - xj0, r1 = xi1 - xj1, r2 = xi2 - xj2;
      float u0 = uvec[0], u1 = uvec[1], u2 = uvec[2];
      s0 = r0 * r0 + r1 * r1 + r2 * r2;
      s1 = xi0 * u0 + xi1 * u1 + xi2 * u2;
      s2 = xj0 * u0 + xj1 * u1 + xj2 * u2;
      s3 = r0 * u0 + r1 * u1 + r2 * u2;
    }
    half_t* rp = sA + (size_t)e * KP1;
    rp[192] = (half_t)s0; rp[193] = (half_t)s1;
    rp[194] = (half_t)s2; rp[195] = (half_t)s3;
    for (int k = 196; k < KP1; ++k) rp[k] = (half_t)0.f;
  }
  __syncthreads();

  const int lane = tid & 31, wave = tid >> 5;
  const size_t mb = (size_t)wave * 32;   // 32 rows per wave

  // h1 = gelu(edge_input @ ew1 + eb1)
  gemm_tile2(sA + mb * KP1, KP1, KP1, ew1t, eb1, sB + mb * MC_DIM, MC_DIM, MC_DIM, true, lane);
  __syncthreads();
  // h2 = gelu(h1 @ ew2 + eb2)   (reuse sA with stride 128)
  gemm_tile2(sB + mb * MC_DIM, MC_DIM, MC_DIM, ew2t, eb2, sA + mb * MC_DIM, MC_DIM, MC_DIM, true, lane);
  __syncthreads();
  // h3 = h2 @ ew3 + eb3         (no activation; overwrite sB)
  gemm_tile2(sA + mb * MC_DIM, MC_DIM, MC_DIM, ew3t, eb3, sB + mb * MC_DIM, MC_DIM, MC_DIM, false, lane);
  __syncthreads();

  // gate = sigmoid(gelu(h3 @ gw1 + gb1) . gw2 + gb2)  -- never materialize g1
  {
    const half_t* aW = sB + mb * MC_DIM;
    const int r = lane & 15, hi = lane >> 4;
    float ps0[8] = {0.f, 0.f, 0.f, 0.f, 0.f, 0.f, 0.f, 0.f};
    float ps1[8] = {0.f, 0.f, 0.f, 0.f, 0.f, 0.f, 0.f, 0.f};
    for (int nt = 0; nt < (MC_DIM >> 4); ++nt) {
      v8f acc0 = {0.f, 0.f, 0.f, 0.f, 0.f, 0.f, 0.f, 0.f};
      v8f acc1 = {0.f, 0.f, 0.f, 0.f, 0.f, 0.f, 0.f, 0.f};
      for (int k = 0; k < (MC_DIM >> 5); ++k) {
        v16h b  = ld_fragB(gw1t, MC_DIM, nt << 4, k << 5, lane);
        v16h a0 = ld_fragA(aW, MC_DIM, k << 5, lane);
        v16h a1 = ld_fragA(aW + (size_t)16 * MC_DIM, MC_DIM, k << 5, lane);
        acc0 = __builtin_amdgcn_wmma_f32_16x16x32_f16(false, a0, false, b,
                                                      (short)0, acc0, false, false);
        acc1 = __builtin_amdgcn_wmma_f32_16x16x32_f16(false, a1, false, b,
                                                      (short)0, acc1, false, false);
      }
      const int n = (nt << 4) + r;
      const float bv = gb1[n], w2 = gw2[n];
#pragma unroll
      for (int v = 0; v < 8; ++v) {
        ps0[v] += gelu_f(acc0[v] + bv) * w2;
        ps1[v] += gelu_f(acc1[v] + bv) * w2;
      }
    }
    const float gb2v = gb2[0];
#pragma unroll
    for (int v = 0; v < 8; ++v) {
      float t0 = ps0[v], t1 = ps1[v];
      for (int s = 1; s < 16; s <<= 1) {
        t0 += __shfl_xor(t0, s, 32);
        t1 += __shfl_xor(t1, s, 32);
      }
      if (r == 0) {
        int e = (int)mb + v + (hi << 3);
        sGate[e]      = 1.0f / (1.0f + expf(-(t0 + gb2v)));
        sGate[e + 16] = 1.0f / (1.0f + expf(-(t1 + gb2v)));
      }
    }
  }
  __syncthreads();

  // Scatter msg = gate * h3 into agg[row] with hardware f32 atomics (L2-resident)
  for (int i = tid; i < TILE_M * MC_DIM; i += NTHREAD) {
    int e = i >> 7, c = i & 127;
    int r = sRow[e];
    if (r >= 0) {
      float val = sGate[e] * (float)sB[(size_t)e * MC_DIM + c];
      atom_add_f32(&agg[(size_t)r * MC_DIM + c], val);
    }
  }
  if (sRow[tid] >= 0) atom_add_f32(&deg[sRow[tid]], 1.0f);
}

// ---------------------------------------------------------------------------
// Update kernel: 64 nodes per workgroup (2 waves x 32 rows); f32 output
// ---------------------------------------------------------------------------
__global__ __launch_bounds__(NTHREAD) void update_kernel(
    const half_t* __restrict__ feat, const float* __restrict__ agg,
    const float* __restrict__ deg, const half_t* __restrict__ gctx,
    const half_t* __restrict__ uw1t, const float* __restrict__ ub1,
    const half_t* __restrict__ uw2t, const float* __restrict__ ub2,
    const half_t* __restrict__ uw3t, const float* __restrict__ ub3,
    const float* __restrict__ res_scale, float* __restrict__ out, int N) {
  __shared__ __align__(16) half_t sA[TILE_M * UPD_KP];   // input, later h2
  __shared__ __align__(16) half_t sB[TILE_M * MC_DIM];   // h1

  const int tid = threadIdx.x;
  const int n0 = blockIdx.x * TILE_M;

  // Build update_input [64 x 256] = [feat | agg/deg | global_ctx]
  for (int i = tid; i < TILE_M * C_DIM; i += NTHREAD) {
    int e = i >> 6, c = i & 63;
    int node = n0 + e;
    half_t* rp = sA + (size_t)e * UPD_KP;
    rp[c] = (node < N) ? feat[(size_t)node * C_DIM + c] : (half_t)0.f;
    rp[C_DIM + MC_DIM + c] = (node < N) ? gctx[c] : (half_t)0.f;
  }
  for (int i = tid; i < TILE_M * MC_DIM; i += NTHREAD) {
    int e = i >> 7, c = i & 127;
    int node = n0 + e;
    float v = 0.f;
    if (node < N) {
      float d = deg[node]; d = (d > 1.f) ? d : 1.f;
      v = agg[(size_t)node * MC_DIM + c] / d;
    }
    sA[(size_t)e * UPD_KP + C_DIM + c] = (half_t)v;
  }
  __syncthreads();

  const int lane = tid & 31, wave = tid >> 5;
  const size_t mb = (size_t)wave * 32;

  // h1 = gelu(in @ uw1 + ub1)
  gemm_tile2(sA + mb * UPD_KP, UPD_KP, UPD_KP, uw1t, ub1, sB + mb * MC_DIM, MC_DIM, MC_DIM, true, lane);
  __syncthreads();
  // h2 = gelu(h1 @ uw2 + ub2)   (reuse sA with stride 128)
  gemm_tile2(sB + mb * MC_DIM, MC_DIM, MC_DIM, uw2t, ub2, sA + mb * MC_DIM, MC_DIM, MC_DIM, true, lane);
  __syncthreads();

  // dh = h2 @ uw3 + ub3; out = tanh(res_scale) * dh   (f32 store)
  {
    const half_t* aW = sA + mb * MC_DIM;
    const float ts = tanhf(res_scale[0]);
    const int r = lane & 15, hi = lane >> 4;
    for (int nt = 0; nt < (C_DIM >> 4); ++nt) {
      v8f acc0 = {0.f, 0.f, 0.f, 0.f, 0.f, 0.f, 0.f, 0.f};
      v8f acc1 = {0.f, 0.f, 0.f, 0.f, 0.f, 0.f, 0.f, 0.f};
      for (int k = 0; k < (MC_DIM >> 5); ++k) {
        v16h b  = ld_fragB(uw3t, MC_DIM, nt << 4, k << 5, lane);
        v16h a0 = ld_fragA(aW, MC_DIM, k << 5, lane);
        v16h a1 = ld_fragA(aW + (size_t)16 * MC_DIM, MC_DIM, k << 5, lane);
        acc0 = __builtin_amdgcn_wmma_f32_16x16x32_f16(false, a0, false, b,
                                                      (short)0, acc0, false, false);
        acc1 = __builtin_amdgcn_wmma_f32_16x16x32_f16(false, a1, false, b,
                                                      (short)0, acc1, false, false);
      }
      const int n = (nt << 4) + r;
      const float bv = ub3[n];
#pragma unroll
      for (int v = 0; v < 8; ++v) {
        int node0 = n0 + (int)mb + v + (hi << 3);
        int node1 = node0 + 16;
        if (node0 < N) out[(size_t)node0 * C_DIM + n] = ts * (acc0[v] + bv);
        if (node1 < N) out[(size_t)node1 * C_DIM + n] = ts * (acc1[v] + bv);
      }
    }
  }
}

// ---------------------------------------------------------------------------
// Host launch
// ---------------------------------------------------------------------------
extern "C" void kernel_launch(void* const* d_in, const int* in_sizes, int n_in,
                              void* d_out, int out_size, void* d_ws, size_t ws_size,
                              hipStream_t stream) {
  (void)n_in; (void)out_size; (void)ws_size;
  const float* coords = (const float*)d_in[0];
  const float* h      = (const float*)d_in[1];
  const float* flow   = (const float*)d_in[2];
  const int*   eidx   = (const int*)d_in[3];
  const float* ln_g   = (const float*)d_in[4];
  const float* ln_b   = (const float*)d_in[5];
  const float* ew1 = (const float*)d_in[6];  const float* eb1 = (const float*)d_in[7];
  const float* ew2 = (const float*)d_in[8];  const float* eb2 = (const float*)d_in[9];
  const float* ew3 = (const float*)d_in[10]; const float* eb3 = (const float*)d_in[11];
  const float* gw1 = (const float*)d_in[12]; const float* gb1 = (const float*)d_in[13];
  const float* gw2 = (const float*)d_in[14]; const float* gb2 = (const float*)d_in[15];
  const float* glw1 = (const float*)d_in[16]; const float* glb1 = (const float*)d_in[17];
  const float* glw2 = (const float*)d_in[18]; const float* glb2 = (const float*)d_in[19];
  const float* glw3 = (const float*)d_in[20]; const float* glb3 = (const float*)d_in[21];
  const float* uw1 = (const float*)d_in[22]; const float* ub1 = (const float*)d_in[23];
  const float* uw2 = (const float*)d_in[24]; const float* ub2 = (const float*)d_in[25];
  const float* uw3 = (const float*)d_in[26]; const float* ub3 = (const float*)d_in[27];
  const float* res_scale = (const float*)d_in[28];

  const int N = in_sizes[1] / C_DIM;
  const int E = in_sizes[3] / 2;
  const int* rows = eidx;
  const int* cols = eidx + E;

  // Workspace layout (256B aligned)
  char* ws = (char*)d_ws;
  size_t off = 0;
  auto alloc = [&](size_t bytes) { size_t r = off; off = (off + bytes + 255) & ~(size_t)255; return r; };
  float*  agg    = (float*)(ws + alloc((size_t)N * MC_DIM * sizeof(float)));
  float*  deg    = (float*)(ws + alloc((size_t)N * sizeof(float)));
  half_t* feat   = (half_t*)(ws + alloc((size_t)N * C_DIM * sizeof(half_t)));
  float*  fsum   = (float*)(ws + alloc(C_DIM * sizeof(float)));
  float*  uvec   = (float*)(ws + alloc(4 * sizeof(float)));
  half_t* gctx   = (half_t*)(ws + alloc(C_DIM * sizeof(half_t)));
  half_t* ew1t   = (half_t*)(ws + alloc((size_t)MC_DIM * KP1 * sizeof(half_t)));
  half_t* ew2t   = (half_t*)(ws + alloc((size_t)MC_DIM * MC_DIM * sizeof(half_t)));
  half_t* ew3t   = (half_t*)(ws + alloc((size_t)MC_DIM * MC_DIM * sizeof(half_t)));
  half_t* gw1t   = (half_t*)(ws + alloc((size_t)MC_DIM * MC_DIM * sizeof(half_t)));
  half_t* uw1t   = (half_t*)(ws + alloc((size_t)MC_DIM * UPD_KP * sizeof(half_t)));
  half_t* uw2t   = (half_t*)(ws + alloc((size_t)MC_DIM * MC_DIM * sizeof(half_t)));
  half_t* uw3t   = (half_t*)(ws + alloc((size_t)C_DIM * MC_DIM * sizeof(half_t)));

  hipMemsetAsync(agg, 0, (size_t)N * MC_DIM * sizeof(float), stream);
  hipMemsetAsync(deg, 0, (size_t)N * sizeof(float), stream);
  hipMemsetAsync(fsum, 0, C_DIM * sizeof(float), stream);

  auto tl = [&](const float* w, half_t* wt, int K, int Ncols, int Kpad) {
    int cnt = Ncols * Kpad;
    wprep_kernel<<<(cnt + 255) / 256, 256, 0, stream>>>(w, wt, K, Ncols, Kpad);
  };
  tl(ew1, ew1t, 3 * C_DIM + 4, MC_DIM, KP1);
  tl(ew2, ew2t, MC_DIM, MC_DIM, MC_DIM);
  tl(ew3, ew3t, MC_DIM, MC_DIM, MC_DIM);
  tl(gw1, gw1t, MC_DIM, MC_DIM, MC_DIM);
  tl(uw1, uw1t, UPD_KP, MC_DIM, UPD_KP);
  tl(uw2, uw2t, MC_DIM, MC_DIM, MC_DIM);
  tl(uw3, uw3t, MC_DIM, C_DIM, MC_DIM);

  unorm_kernel<<<1, 1, 0, stream>>>(flow, uvec);
  ln_kernel<<<(N + 255) / 256, 256, 0, stream>>>(h, ln_g, ln_b, feat, fsum, N);
  global_mlp_kernel<<<1, C_DIM, 0, stream>>>(fsum, N, glw1, glb1, glw2, glb2, glw3, glb3, gctx);

  edge_kernel<<<(E + TILE_M - 1) / TILE_M, NTHREAD, 0, stream>>>(
      rows, cols, coords, feat, uvec,
      ew1t, eb1, ew2t, eb2, ew3t, eb3, gw1t, gb1, gw2, gb2, agg, deg, E);

  update_kernel<<<(N + TILE_M - 1) / TILE_M, NTHREAD, 0, stream>>>(
      feat, agg, deg, gctx, uw1t, ub1, uw2t, ub2, uw3t, ub3,
      res_scale, (float*)d_out, N);
}
